// EfficientSelfAttention_70789650972784
// MI455X (gfx1250) — compile-verified, matrix operand order checked by reading, not run on hardware
//
#include <hip/hip_runtime.h>
#include <hip/hip_bf16.h>

typedef _Float16 half_t;
typedef __attribute__((ext_vector_type(16))) _Float16 v16h;
typedef __attribute__((ext_vector_type(8)))  float    v8f;

#define BATCH  4
#define CDIM   512
#define HEADS  8
#define HD     64
#define NPOS   4096   // 64*64
#define NKPOS  1024   // 32*32
#define ATT_SCALE 0.125f

// ---------------------------------------------------------------------------
// WMMA helpers.  16-bit A-matrix 16x32 layout (ISA 7.12.2): lane L holds row
// m = L%16; its 16 halves are two contiguous 8-half chunks at k = off..off+7
// and k = off+16..off+23, off = (L>=16 ? 8 : 0).  B uses the symmetric layout
// (lane = column, identical k packing), so both fragments are two 16-byte
// loads from a k-contiguous row.
// ---------------------------------------------------------------------------
__device__ __forceinline__ v16h frag_ld(const half_t* rowbase, int lane) {
  const int off = (lane & 16) ? 8 : 0;
  union { uint4 u[2]; v16h v; } t;
  t.u[0] = *(const uint4*)(rowbase + off);
  t.u[1] = *(const uint4*)(rowbase + off + 16);
  return t.v;
}

__device__ __forceinline__ v8f wmma16(v16h a, v16h b, v8f c) {
  return __builtin_amdgcn_wmma_f32_16x16x32_f16(false, a, false, b,
                                                (short)0, c, false, false);
}

// Low 32 bits of a flat pointer into LDS == LDS byte offset (ISA 10.2:
// the LDS aperture maps by truncating to addr[31:0]).
__device__ __forceinline__ uint32_t lds_addr32(const void* p) {
  return (uint32_t)(uintptr_t)p;
}

// async global->LDS, 16B per lane, GVS addressing (SGPR base + VGPR offset).
// Tracked with ASYNCcnt; no VGPR destination, no LOADcnt/DScnt involvement.
__device__ __forceinline__ void async_ld_b128(uint32_t lds_off, uint32_t goff,
                                              unsigned long long base) {
  asm volatile("global_load_async_to_lds_b128 %0, %1, %2"
               :: "v"(lds_off), "v"(goff), "s"(base) : "memory");
}

// ---------------------------------------------------------------------------
// Conversion kernels (f32 -> f16, layout shuffles)
// ---------------------------------------------------------------------------
__global__ void cvt_f16(const float* __restrict__ in, half_t* __restrict__ out, int n) {
  int i = blockIdx.x * 256 + threadIdx.x;
  if (i < n) out[i] = (half_t)in[i];
}

// x [B][512][4096] -> xT [B][4096][512] (f16, position-major)
__global__ void cvt_xT(const float* __restrict__ x, half_t* __restrict__ xT) {
  int i = blockIdx.x * 256 + threadIdx.x;        // over B*512*4096
  int b = i >> 21;
  int rem = i & ((1 << 21) - 1);
  int c = rem >> 12;
  int n = rem & 4095;
  xT[((size_t)b * NPOS + n) * CDIM + c] = (half_t)x[i];
}

// sr_w [O=512][C=512][2][2] -> srwh [o][k], k = (sy*2+sx)*512 + c
__global__ void cvt_srw(const float* __restrict__ srw, half_t* __restrict__ out) {
  int i = blockIdx.x * 256 + threadIdx.x;        // over 512*2048
  int o = i >> 11;
  int k = i & 2047;
  int s = k >> 9;
  int c = k & 511;
  int sy = s >> 1, sx = s & 1;
  out[i] = (half_t)srw[(((size_t)o * 512 + c) * 2 + sy) * 2 + sx];
}

// ---------------------------------------------------------------------------
// LayerNorm over channels at each reduced spatial position.
// kvin [B*1024][512] f32 -> lnT [B*1024][512] f16
// ---------------------------------------------------------------------------
__global__ __launch_bounds__(256)
void ln_kernel(const float* __restrict__ kvin, const float* __restrict__ g,
               const float* __restrict__ be, half_t* __restrict__ lnT) {
  __shared__ float r1[256], r2[256];
  const int pos = blockIdx.x;
  const int t = threadIdx.x;
  const float* xr = kvin + (size_t)pos * CDIM;
  float x0 = xr[t], x1 = xr[t + 256];
  r1[t] = x0 + x1;
  r2[t] = x0 * x0 + x1 * x1;
  __syncthreads();
  for (int off = 128; off > 0; off >>= 1) {
    if (t < off) { r1[t] += r1[t + off]; r2[t] += r2[t + off]; }
    __syncthreads();
  }
  float mean = r1[0] * (1.0f / 512.0f);
  float var  = r2[0] * (1.0f / 512.0f) - mean * mean;
  float rs   = rsqrtf(var + 1e-6f);
  lnT[(size_t)pos * CDIM + t]       = (half_t)((x0 - mean) * rs * g[t]       + be[t]);
  lnT[(size_t)pos * CDIM + t + 256] = (half_t)((x1 - mean) * rs * g[t + 256] + be[t + 256]);
}

// ---------------------------------------------------------------------------
// Tiled WMMA GEMM: out[M][Npos] = W[M][K] * act[Npos][K]^T  (+bias)
// Block = 256 threads = 8 waves, 128x128 tile, k-step 32.
// Tiles are staged with double-buffered GLOBAL_LOAD_ASYNC_TO_LDS_B128:
// tile t+1 is fetched (ASYNCcnt) while tile t runs through the WMMAs.
// MODE 0: Q proj   K=512 NP=4096 -> qh  [bh][n][d]   f16
// MODE 1: SR conv  K=2048 NP=1024 (im2col gather from xT) -> kvin f32 [b][n][o]
// MODE 2: KV proj  K=512 NP=1024 -> kh [bh][nk][d], vt [bh][d][nk]  f16
// MODE 3: out proj K=512 NP=4096 -> d_out f32 [b][o][n]
// ---------------------------------------------------------------------------
template<int MODE>
__global__ __launch_bounds__(256)
void gemm_wmma(const half_t* __restrict__ W, const half_t* __restrict__ actBase,
               const float* __restrict__ bias,
               float* __restrict__ outF, half_t* __restrict__ outH,
               half_t* __restrict__ outH2) {
  constexpr int K  = (MODE == 1) ? 2048 : 512;
  constexpr int TILE = 128 * 40;    // halves per buffer, 40-half pitch (conflict free)

  __shared__ half_t Ah[2 * TILE];
  __shared__ half_t Bh[2 * TILE];

  const int b    = blockIdx.z;
  const int n0   = blockIdx.x * 128;
  const int m0   = blockIdx.y * 128;
  const int tid  = threadIdx.x;
  const int lane = tid & 31;
  const int wave = tid >> 5;
  const int wm   = wave & 3;   // 4 row blocks of 32
  const int wn   = wave >> 2;  // 2 col blocks of 64
  const half_t* act = actBase + (size_t)b * ((MODE == 2) ? (NKPOS * CDIM) : (NPOS * CDIM));

  const uint32_t ldsA = lds_addr32(&Ah[0]);
  const uint32_t ldsB = lds_addr32(&Bh[0]);
  const unsigned long long wbase = (unsigned long long)(uintptr_t)W;
  const unsigned long long abase = (unsigned long long)(uintptr_t)act;

  // Issue one 128x32 A tile + B tile into buffer `buf` (4 async instrs / wave).
  auto issue_tile = [&](int buf, int k0) {
    #pragma unroll
    for (int i = 0; i < 2; i++) {
      int idx = tid + i * 256;
      int row = idx >> 2;
      int ch  = (idx & 3) * 8;           // 8-half (16B) chunk
      uint32_t lofs = (uint32_t)(buf * TILE + row * 40 + ch) * 2u;
      async_ld_b128(ldsA + lofs, (uint32_t)(((m0 + row) * K + k0 + ch) * 2), wbase);
      uint32_t gb;
      if (MODE == 1) {
        int kk = k0 + ch;
        int s  = kk >> 9;                // 2x2 tap selector
        int c  = kk & 511;
        int sy = s >> 1, sx = s & 1;
        int n  = n0 + row;               // reduced position
        int py = n >> 5, px = n & 31;
        int sp = (2 * py + sy) * 64 + (2 * px + sx);
        gb = (uint32_t)((sp * CDIM + c) * 2);
      } else {
        gb = (uint32_t)(((n0 + row) * K + k0 + ch) * 2);
      }
      async_ld_b128(ldsB + lofs, gb, abase);
    }
  };

  v8f acc[2][4];
  const v8f vzero = {0.f, 0.f, 0.f, 0.f, 0.f, 0.f, 0.f, 0.f};
  for (int i = 0; i < 2; i++)
    for (int j = 0; j < 4; j++) acc[i][j] = vzero;

  issue_tile(0, 0);
  for (int k0 = 0, it = 0; k0 < K; k0 += 32, ++it) {
    const int cur = it & 1;
    const bool more = (k0 + 32) < K;
    if (more) issue_tile(cur ^ 1, k0 + 32);
    // async loads complete in order: <=4 outstanding means current tile landed
    if (more) asm volatile("s_wait_asynccnt 0x4" ::: "memory");
    else      asm volatile("s_wait_asynccnt 0x0" ::: "memory");
    __syncthreads();                     // every wave's part of the tile is in LDS

    v16h af[2], bf[4];
    #pragma unroll
    for (int mt = 0; mt < 2; mt++)
      af[mt] = frag_ld(&Ah[cur * TILE + (wm * 32 + mt * 16 + (lane & 15)) * 40], lane);
    #pragma unroll
    for (int nt = 0; nt < 4; nt++)
      bf[nt] = frag_ld(&Bh[cur * TILE + (wn * 64 + nt * 16 + (lane & 15)) * 40], lane);
    #pragma unroll
    for (int mt = 0; mt < 2; mt++)
      #pragma unroll
      for (int nt = 0; nt < 4; nt++)
        acc[mt][nt] = wmma16(af[mt], bf[nt], acc[mt][nt]);

    // fragment reads done before anyone overwrites this buffer
    asm volatile("s_wait_dscnt 0x0" ::: "memory");
    __syncthreads();
  }

  // epilogue: C/D layout -> lane holds col n = lane%16, rows r + (lane>=16?8:0)
  const int halfoff = (lane & 16) ? 8 : 0;
  const int ncol = lane & 15;
  #pragma unroll
  for (int mt = 0; mt < 2; mt++) {
    #pragma unroll
    for (int nt = 0; nt < 4; nt++) {
      #pragma unroll
      for (int r = 0; r < 8; r++) {
        int o = m0 + wm * 32 + mt * 16 + r + halfoff;
        int n = n0 + wn * 64 + nt * 16 + ncol;
        float v = acc[mt][nt][r] + bias[o];
        if (MODE == 0) {
          int head = o >> 6, d = o & 63;
          outH[(((size_t)(b * HEADS + head) * NPOS) + n) * HD + d] = (half_t)v;
        } else if (MODE == 1) {
          outF[((size_t)b * NKPOS + n) * CDIM + o] = v;
        } else if (MODE == 2) {
          if (o < CDIM) {
            int head = o >> 6, d = o & 63;
            outH[(((size_t)(b * HEADS + head) * NKPOS) + n) * HD + d] = (half_t)v;
          } else {
            int vc = o - CDIM;
            int head = vc >> 6, d = vc & 63;
            outH2[(((size_t)(b * HEADS + head) * HD) + d) * NKPOS + n] = (half_t)v;
          }
        } else { // MODE 3
          outF[((size_t)b * CDIM + o) * NPOS + n] = v;
        }
      }
    }
  }
}

// ---------------------------------------------------------------------------
// Flash attention: 256 threads = 8 waves; each wave owns 16 queries and loops
// over keys in 32-chunks with online softmax.  Q/K/V fragments loaded straight
// from global (layouts are k-contiguous by construction); only the probability
// tile round-trips through per-wave LDS for the C/D -> A-layout transpose.
// ---------------------------------------------------------------------------
__global__ __launch_bounds__(256)
void attn_kernel(const half_t* __restrict__ qh, const half_t* __restrict__ kh,
                 const half_t* __restrict__ vt, half_t* __restrict__ attnT) {
  __shared__ half_t Pbuf[8 * 16 * 40];
  const int tid  = threadIdx.x;
  const int lane = tid & 31;
  const int wave = tid >> 5;
  const int bh   = blockIdx.y;          // b*8 + h
  const int b    = bh >> 3, h = bh & 7;
  const int q0   = blockIdx.x * 128 + wave * 16;
  const int lrow = lane & 15;
  const int halfoff = (lane & 16) ? 8 : 0;
  half_t* Pw = &Pbuf[wave * 16 * 40];

  // Q A-fragments (16 q x 64 d = two k-steps), loaded once
  const half_t* qrow = qh + ((size_t)bh * NPOS + q0 + lrow) * HD;
  v16h qa0 = frag_ld(qrow, lane);
  v16h qa1 = frag_ld(qrow + 32, lane);

  const v8f vzero = {0.f, 0.f, 0.f, 0.f, 0.f, 0.f, 0.f, 0.f};
  v8f acc[4];
  for (int dt = 0; dt < 4; dt++) acc[dt] = vzero;
  float rmax[8], rsum[8];
  for (int r = 0; r < 8; r++) { rmax[r] = -1e30f; rsum[r] = 0.f; }

  for (int kb = 0; kb < NKPOS; kb += 32) {
    // S = (Q K^T) * scale for a 16x32 chunk (two 16x16 tiles)
    v8f s[2];
    #pragma unroll
    for (int st = 0; st < 2; st++) {
      const half_t* kbase = kh + ((size_t)bh * NKPOS + kb + st * 16 + lrow) * HD;
      v16h kf0 = frag_ld(kbase, lane);
      v16h kf1 = frag_ld(kbase + 32, lane);
      v8f z = vzero;
      z = wmma16(qa0, kf0, z);
      z = wmma16(qa1, kf1, z);
      #pragma unroll
      for (int r = 0; r < 8; r++) z[r] *= ATT_SCALE;
      s[st] = z;
    }
    // online softmax, rows live in 16-lane halves
    #pragma unroll
    for (int r = 0; r < 8; r++) {
      float m = fmaxf(s[0][r], s[1][r]);
      for (int off = 1; off < 16; off <<= 1) m = fmaxf(m, __shfl_xor(m, off, 32));
      float nm = fmaxf(rmax[r], m);
      float alpha = __expf(rmax[r] - nm);
      rmax[r] = nm;
      float p0 = __expf(s[0][r] - nm);
      float p1 = __expf(s[1][r] - nm);
      s[0][r] = p0; s[1][r] = p1;
      float ps = p0 + p1;
      for (int off = 1; off < 16; off <<= 1) ps += __shfl_xor(ps, off, 32);
      rsum[r] = rsum[r] * alpha + ps;
      #pragma unroll
      for (int dt = 0; dt < 4; dt++) acc[dt][r] *= alpha;
    }
    // transpose P (C/D layout) into A layout via per-wave LDS tile
    #pragma unroll
    for (int st = 0; st < 2; st++)
      #pragma unroll
      for (int r = 0; r < 8; r++)
        Pw[(r + halfoff) * 40 + st * 16 + lrow] = (half_t)s[st][r];
    asm volatile("s_wait_dscnt 0x0" ::: "memory");
    v16h pa = frag_ld(&Pw[lrow * 40], lane);
    // acc += P (16x32) * V (32 keys x 16 d), V^T layout -> contiguous keys
    #pragma unroll
    for (int dt = 0; dt < 4; dt++) {
      const half_t* vbase = vt + ((size_t)bh * HD + dt * 16 + lrow) * NKPOS + kb;
      v16h vf = frag_ld(vbase, lane);
      acc[dt] = wmma16(pa, vf, acc[dt]);
    }
  }

  // normalize and emit position-major [n][c] f16 for the projection GEMM
  #pragma unroll
  for (int dt = 0; dt < 4; dt++)
    #pragma unroll
    for (int r = 0; r < 8; r++) {
      int orow = q0 + r + halfoff;
      int c = h * HD + dt * 16 + lrow;
      attnT[((size_t)b * NPOS + orow) * CDIM + c] = (half_t)(acc[dt][r] / rsum[r]);
    }
}

// ---------------------------------------------------------------------------
extern "C" void kernel_launch(void* const* d_in, const int* in_sizes, int n_in,
                              void* d_out, int out_size, void* d_ws, size_t ws_size,
                              hipStream_t stream) {
  const float* x      = (const float*)d_in[0];
  const float* q_w    = (const float*)d_in[1];
  const float* q_b    = (const float*)d_in[2];
  const float* kv_w   = (const float*)d_in[3];
  const float* kv_b   = (const float*)d_in[4];
  const float* sr_w   = (const float*)d_in[5];
  const float* sr_b   = (const float*)d_in[6];
  const float* ln_g   = (const float*)d_in[7];
  const float* ln_b   = (const float*)d_in[8];
  const float* proj_w = (const float*)d_in[9];
  const float* proj_b = (const float*)d_in[10];
  float* out = (float*)d_out;

  size_t off = 0;
  auto carve = [&](size_t bytes) {
    void* p = (char*)d_ws + off;
    off += (bytes + 255) & ~(size_t)255;
    return p;
  };
  half_t* xT    = (half_t*)carve((size_t)BATCH * NPOS * CDIM * 2);
  half_t* qwh   = (half_t*)carve((size_t)CDIM * CDIM * 2);
  half_t* kvwh  = (half_t*)carve((size_t)2 * CDIM * CDIM * 2);
  half_t* srwh  = (half_t*)carve((size_t)CDIM * 2048 * 2);
  half_t* pwh   = (half_t*)carve((size_t)CDIM * CDIM * 2);
  half_t* qhb   = (half_t*)carve((size_t)BATCH * HEADS * NPOS * HD * 2);
  float*  kvin  = (float*)carve((size_t)BATCH * NKPOS * CDIM * 4);
  half_t* lnT   = (half_t*)carve((size_t)BATCH * NKPOS * CDIM * 2);
  half_t* khb   = (half_t*)carve((size_t)BATCH * HEADS * NKPOS * HD * 2);
  half_t* vtb   = (half_t*)carve((size_t)BATCH * HEADS * HD * NKPOS * 2);
  half_t* attnT = (half_t*)carve((size_t)BATCH * NPOS * CDIM * 2);

  // 1. precision / layout conversions
  cvt_xT<<<(BATCH * CDIM * NPOS) / 256, 256, 0, stream>>>(x, xT);
  cvt_f16<<<(CDIM * CDIM) / 256, 256, 0, stream>>>(q_w, qwh, CDIM * CDIM);
  cvt_f16<<<(2 * CDIM * CDIM) / 256, 256, 0, stream>>>(kv_w, kvwh, 2 * CDIM * CDIM);
  cvt_f16<<<(CDIM * CDIM) / 256, 256, 0, stream>>>(proj_w, pwh, CDIM * CDIM);
  cvt_srw<<<(CDIM * 2048) / 256, 256, 0, stream>>>(sr_w, srwh);

  // 2. SR conv as im2col GEMM -> kvin f32 [b][n][c]
  gemm_wmma<1><<<dim3(NKPOS / 128, CDIM / 128, BATCH), 256, 0, stream>>>(
      srwh, xT, sr_b, kvin, nullptr, nullptr);

  // 3. channel LayerNorm -> lnT f16
  ln_kernel<<<BATCH * NKPOS, 256, 0, stream>>>(kvin, ln_g, ln_b, lnT);

  // 4. KV projection -> kh [bh][nk][d], vt [bh][d][nk]
  gemm_wmma<2><<<dim3(NKPOS / 128, (2 * CDIM) / 128, BATCH), 256, 0, stream>>>(
      kvwh, lnT, kv_b, nullptr, khb, vtb);

  // 5. Q projection -> qh [bh][n][d]
  gemm_wmma<0><<<dim3(NPOS / 128, CDIM / 128, BATCH), 256, 0, stream>>>(
      qwh, xT, q_b, nullptr, qhb, nullptr);

  // 6. flash attention -> attnT f16 [b][n][c]
  attn_kernel<<<dim3(NPOS / 128, BATCH * HEADS), 256, 0, stream>>>(
      qhb, khb, vtb, attnT);

  // 7. output projection -> d_out f32 [b][c][n]
  gemm_wmma<3><<<dim3(NPOS / 128, CDIM / 128, BATCH), 256, 0, stream>>>(
      pwh, attnT, proj_b, out, nullptr, nullptr);
}
